// CGC_OPE_72688026518219
// MI455X (gfx1250) — matrix operand order repaired and checked by reading, not compiled, not run on hardware
//
#include <hip/hip_runtime.h>
#include <stdint.h>
#include <stddef.h>

// ---------------------------------------------------------------------------
// Problem constants
// ---------------------------------------------------------------------------
#define B_TOK 4096
#define D_FULL 1024
#define D_TASK 512
#define HID 512
#define OUTC 256
#define NEXP 4
#define LPAD 40   // LDS row stride (ushorts) for 32-K tiles: 80B, 20-bank stride
#define LP64 72   // LDS row stride (ushorts) for 64-K tiles: 144B, 36-bank stride

typedef __attribute__((ext_vector_type(16))) __bf16 v16bf;
typedef __attribute__((ext_vector_type(8)))  __bf16 v8bf;
typedef __attribute__((ext_vector_type(8)))  float  v8f;

// ---------------------------------------------------------------------------
// Helpers
// ---------------------------------------------------------------------------
__device__ __forceinline__ unsigned short f32_to_bf16_bits(float f) {
  unsigned int u = __float_as_uint(f);
  u += 0x7FFFu + ((u >> 16) & 1u);       // round-to-nearest-even
  return (unsigned short)(u >> 16);
}

// CDNA5 async global->LDS copy, 16 bytes per lane. Tracked by ASYNCcnt.
__device__ __forceinline__ void async_cp16(unsigned short* lds_p,
                                           const unsigned short* g_p) {
  uint32_t l = (uint32_t)(uintptr_t)lds_p;   // low 32 bits = LDS offset
  uint64_t g = (uint64_t)(uintptr_t)g_p;
  asm volatile("global_load_async_to_lds_b128 %0, %1, off"
               :: "v"(l), "v"(g) : "memory");
}

__device__ __forceinline__ void wait_async0() {
  asm volatile("s_wait_asynccnt 0x0" ::: "memory");
}
__device__ __forceinline__ void wait_async4() {
  asm volatile("s_wait_asynccnt 0x4" ::: "memory");
}
__device__ __forceinline__ void wait_async6() {
  asm volatile("s_wait_asynccnt 0x6" ::: "memory");
}

// A-fragment (16x32 bf16) from LDS: contiguous 32B run: chunks +0 / +8.
__device__ __forceinline__ v16bf lds_a_frag(const unsigned short* p) {
  union { v16bf v; v8bf h[2]; } u;
  u.h[0] = *(const v8bf*)(p);
  u.h[1] = *(const v8bf*)(p + 8);
  return u.v;
}
// B-fragment (32x16 bf16) from LDS: contiguous 32B run: chunks +0 / +8.
__device__ __forceinline__ v16bf lds_b_frag(const unsigned short* p) {
  union { v16bf v; v8bf h[2]; } u;
  u.h[0] = *(const v8bf*)(p);
  u.h[1] = *(const v8bf*)(p + 8);
  return u.v;
}

#define WMMA_BF16(A, Bf, C) \
  __builtin_amdgcn_wmma_f32_16x16x32_bf16(false, (A), false, (Bf), (short)0, (C), false, false)

// --- 32-K staging (GEMM1) ---
__device__ __forceinline__ void stage128(unsigned short* ldsb,
                                         const unsigned short* g, int D,
                                         int tid) {
  int r0 = tid >> 2, p0 = (tid & 3) * 8;
  async_cp16(ldsb + r0 * LPAD + p0, g + (size_t)r0 * D + p0);
  async_cp16(ldsb + (r0 + 64) * LPAD + p0, g + (size_t)(r0 + 64) * D + p0);
}

// --- 64-K staging (GEMM2) ---
__device__ __forceinline__ void stage128x64(unsigned short* ldsb,
                                            const unsigned short* g, int D,
                                            int tid) {
  #pragma unroll
  for (int i = 0; i < 4; ++i) {
    int c = tid + i * 256;
    int row = c >> 3, part = (c & 7) * 8;
    async_cp16(ldsb + row * LP64 + part, g + (size_t)row * D + part);
  }
}
__device__ __forceinline__ void stage64x64(unsigned short* ldsb,
                                           const unsigned short* g, int D,
                                           int tid) {
  #pragma unroll
  for (int i = 0; i < 2; ++i) {
    int c = tid + i * 256;
    int row = c >> 3, part = (c & 7) * 8;
    async_cp16(ldsb + row * LP64 + part, g + (size_t)row * D + part);
  }
}

// ---------------------------------------------------------------------------
// f32 -> bf16 convert (8 elems/thread, 16B stores)
// ---------------------------------------------------------------------------
__global__ void cvt_bf16_kernel(const float* __restrict__ in,
                                unsigned short* __restrict__ out, int n8) {
  int i = blockIdx.x * blockDim.x + threadIdx.x;
  if (i >= n8) return;
  const float4* p = (const float4*)(in + (size_t)i * 8);
  float4 f0 = p[0], f1 = p[1];
  union { unsigned short s[8]; uint4 v; } u;
  u.s[0] = f32_to_bf16_bits(f0.x); u.s[1] = f32_to_bf16_bits(f0.y);
  u.s[2] = f32_to_bf16_bits(f0.z); u.s[3] = f32_to_bf16_bits(f0.w);
  u.s[4] = f32_to_bf16_bits(f1.x); u.s[5] = f32_to_bf16_bits(f1.y);
  u.s[6] = f32_to_bf16_bits(f1.z); u.s[7] = f32_to_bf16_bits(f1.w);
  *(uint4*)(out + (size_t)i * 8) = u.v;
}

// f32 [E,R,C] -> bf16 [E,C,R]: each thread emits an 8-run of the output row
// (one 16B store); its 8 input reads are lane-coalesced along C.
__global__ void cvt_tr_bf16_kernel(const float* __restrict__ in,
                                   unsigned short* __restrict__ out,
                                   int E, int R, int C) {
  int i = blockIdx.x * blockDim.x + threadIdx.x;
  int nR8 = R >> 3;
  int total = E * C * nR8;
  if (i >= total) return;
  int c  = i % C;
  int r8 = (i / C) % nR8;
  int e  = i / (C * nR8);
  int r0 = r8 * 8;
  const float* ip = in + (size_t)e * R * C + (size_t)r0 * C + c;
  union { unsigned short s[8]; uint4 v; } u;
  #pragma unroll
  for (int k = 0; k < 8; ++k) u.s[k] = f32_to_bf16_bits(ip[(size_t)k * C]);
  *(uint4*)(out + (size_t)e * C * R + (size_t)c * R + r0) = u.v;
}

// ---------------------------------------------------------------------------
// Gates: one WAVE per token; lanes stride D (coalesced), shfl-xor reduce.
// ---------------------------------------------------------------------------
__device__ __forceinline__ float wave_sum(float v) {
  v += __shfl_xor(v, 16, 32);
  v += __shfl_xor(v, 8, 32);
  v += __shfl_xor(v, 4, 32);
  v += __shfl_xor(v, 2, 32);
  v += __shfl_xor(v, 1, 32);
  return v;
}

__device__ __forceinline__ void softmax_store(float* acc, int n, float* dst) {
  float m = acc[0];
  for (int i = 1; i < n; ++i) m = acc[i] > m ? acc[i] : m;
  float s = 0.f;
  for (int i = 0; i < n; ++i) { float e = __expf(acc[i] - m); acc[i] = e; s += e; }
  float inv = 1.f / s;
  for (int i = 0; i < n; ++i) dst[i] = acc[i] * inv;
}

__global__ __launch_bounds__(256) void gates_kernel(
    const float* __restrict__ xf, const float* __restrict__ x1,
    const float* __restrict__ x2,
    const float* __restrict__ gshW, const float* __restrict__ gshb,
    const float* __restrict__ g1W, const float* __restrict__ g1b,
    const float* __restrict__ g2W, const float* __restrict__ g2b,
    float* __restrict__ gsh, float* __restrict__ g1, float* __restrict__ g2) {
  const int wid  = (blockIdx.x * blockDim.x + threadIdx.x) >> 5;  // token
  const int lane = threadIdx.x & 31;
  if (wid >= B_TOK) return;
  {
    float acc[12] = {};
    const float* xr = xf + (size_t)wid * D_FULL;
    for (int d = lane; d < D_FULL; d += 32) {
      float xv = xr[d];
      const float* w = gshW + (size_t)d * 12;
      #pragma unroll
      for (int j = 0; j < 12; ++j) acc[j] += xv * w[j];
    }
    #pragma unroll
    for (int j = 0; j < 12; ++j) acc[j] = wave_sum(acc[j]) + gshb[j];
    if (lane == 0) softmax_store(acc, 12, gsh + (size_t)wid * 12);
  }
  {
    float acc[8] = {};
    const float* xr = x1 + (size_t)wid * D_TASK;
    for (int d = lane; d < D_TASK; d += 32) {
      float xv = xr[d];
      const float* w = g1W + (size_t)d * 8;
      #pragma unroll
      for (int j = 0; j < 8; ++j) acc[j] += xv * w[j];
    }
    #pragma unroll
    for (int j = 0; j < 8; ++j) acc[j] = wave_sum(acc[j]) + g1b[j];
    if (lane == 0) softmax_store(acc, 8, g1 + (size_t)wid * 8);
  }
  {
    float acc[8] = {};
    const float* xr = x2 + (size_t)wid * D_TASK;
    for (int d = lane; d < D_TASK; d += 32) {
      float xv = xr[d];
      const float* w = g2W + (size_t)d * 8;
      #pragma unroll
      for (int j = 0; j < 8; ++j) acc[j] += xv * w[j];
    }
    #pragma unroll
    for (int j = 0; j < 8; ++j) acc[j] = wave_sum(acc[j]) + g2b[j];
    if (lane == 0) softmax_store(acc, 8, g2 + (size_t)wid * 8);
  }
}

// ---------------------------------------------------------------------------
// GEMM1: h[e] = relu(x @ W1t[e]^T + b1[e]) -> bf16.
// Block tile 128x128; 8 waves of 32x64; 32-K async double buffering.
// Grid: (HID/128, B/128, NEXP).
// ---------------------------------------------------------------------------
__device__ __forceinline__ void store_h_tile(unsigned short* __restrict__ hp,
                                             float bias, v8f c, int rowbase,
                                             int kh, int col) {
  #pragma unroll
  for (int j = 0; j < 8; ++j) {
    float v = c[j] + bias;
    v = v > 0.f ? v : 0.f;
    hp[(size_t)(rowbase + j + kh) * HID + col] = f32_to_bf16_bits(v);
  }
}

__global__ __launch_bounds__(256) void expert_gemm1_kernel(
    const unsigned short* __restrict__ x,    // [B, D] bf16 row-major
    const unsigned short* __restrict__ w1t,  // [E, HID, D] bf16 (transposed)
    const float* __restrict__ b1,            // [E, HID]
    unsigned short* __restrict__ hout,       // [E, B, HID] bf16
    int D) {
  __shared__ unsigned short As[2][128 * LPAD];
  __shared__ unsigned short Bs[2][128 * LPAD];

  const int e    = blockIdx.z;
  const int tid  = threadIdx.x;
  const int lane = tid & 31;
  const int wave = tid >> 5;
  const int l15  = lane & 15;
  const int half = lane >> 4;
  const int kh   = half * 8;
  const int kb   = half * 16;
  const int wm   = (wave >> 1) * 32;
  const int wn   = (wave & 1) * 64;
  const int m0   = blockIdx.y * 128;
  const int n0   = blockIdx.x * 128;

  const unsigned short* ag = x + (size_t)m0 * D;
  const unsigned short* bg = w1t + (size_t)e * HID * D + (size_t)n0 * D;

  v8f c00 = {}, c01 = {}, c02 = {}, c03 = {};
  v8f c10 = {}, c11 = {}, c12 = {}, c13 = {};

  const int nk = D / 32;
  stage128(As[0], ag, D, tid);
  stage128(Bs[0], bg, D, tid);

  for (int kt = 0; kt < nk; ++kt) {
    const int buf = kt & 1;
    if (kt + 1 < nk) {
      stage128(As[buf ^ 1], ag + (kt + 1) * 32, D, tid);
      stage128(Bs[buf ^ 1], bg + (kt + 1) * 32, D, tid);
      wait_async4();
    } else {
      wait_async0();
    }
    __syncthreads();

    const unsigned short* A  = As[buf];
    const unsigned short* Bt = Bs[buf];
    v16bf a0 = lds_a_frag(A + (wm + l15) * LPAD + kb);
    v16bf a1 = lds_a_frag(A + (wm + 16 + l15) * LPAD + kb);
    v16bf b0 = lds_b_frag(Bt + (wn + l15) * LPAD + kb);
    v16bf b1f = lds_b_frag(Bt + (wn + 16 + l15) * LPAD + kb);
    v16bf b2 = lds_b_frag(Bt + (wn + 32 + l15) * LPAD + kb);
    v16bf b3 = lds_b_frag(Bt + (wn + 48 + l15) * LPAD + kb);
    c00 = WMMA_BF16(a0, b0, c00);
    c10 = WMMA_BF16(a1, b0, c10);
    c01 = WMMA_BF16(a0, b1f, c01);
    c11 = WMMA_BF16(a1, b1f, c11);
    c02 = WMMA_BF16(a0, b2, c02);
    c12 = WMMA_BF16(a1, b2, c12);
    c03 = WMMA_BF16(a0, b3, c03);
    c13 = WMMA_BF16(a1, b3, c13);
    __syncthreads();
  }

  unsigned short* hp = hout + (size_t)e * B_TOK * HID;
  const float* bb = b1 + (size_t)e * HID;
  const int rb0 = m0 + wm, rb1 = m0 + wm + 16;
  const int gc0 = n0 + wn + l15, gc1 = gc0 + 16, gc2 = gc0 + 32, gc3 = gc0 + 48;
  store_h_tile(hp, bb[gc0], c00, rb0, kh, gc0);
  store_h_tile(hp, bb[gc0], c10, rb1, kh, gc0);
  store_h_tile(hp, bb[gc1], c01, rb0, kh, gc1);
  store_h_tile(hp, bb[gc1], c11, rb1, kh, gc1);
  store_h_tile(hp, bb[gc2], c02, rb0, kh, gc2);
  store_h_tile(hp, bb[gc2], c12, rb1, kh, gc2);
  store_h_tile(hp, bb[gc3], c03, rb0, kh, gc3);
  store_h_tile(hp, bb[gc3], c13, rb1, kh, gc3);
}

// ---------------------------------------------------------------------------
// GEMM2 + gated combine, fused, deterministic.
// Block tile 128x64; 8 waves of 32x32; 64-K staging (2 WMMA sub-steps per
// barrier pair). Pipeline over (12 experts x 8 phases). Grid: (OUTC/64, B/128).
// ---------------------------------------------------------------------------
__device__ __forceinline__ void accum_tile(
    v8f c, float bias, int col, int rowbase, int kh, int g, int e, int ge,
    const float* __restrict__ gshv, const float* __restrict__ g1v,
    const float* __restrict__ g2v, v8f& ssh, v8f& s1, v8f& s2) {
  #pragma unroll
  for (int j = 0; j < 8; ++j) {
    int row = rowbase + j + kh;
    float v = c[j] + bias;
    v = v > 0.f ? v : 0.f;
    ssh[j] += gshv[(size_t)row * 12 + ge] * v;
    if (g == 0) {
      s1[j] += g1v[(size_t)row * 8 + e] * v;
    } else if (g == 1) {
      s2[j] += g2v[(size_t)row * 8 + e] * v;
    } else {
      s1[j] += g1v[(size_t)row * 8 + 4 + e] * v;
      s2[j] += g2v[(size_t)row * 8 + 4 + e] * v;
    }
  }
}

__global__ __launch_bounds__(256) void moe_gemm2_combine_kernel(
    const unsigned short* __restrict__ h_t1,   // [4, B, HID] bf16
    const unsigned short* __restrict__ h_t2,
    const unsigned short* __restrict__ h_sh,
    const unsigned short* __restrict__ w2_t1,  // [4, OUTC, HID] bf16
    const unsigned short* __restrict__ w2_t2,
    const unsigned short* __restrict__ w2_sh,
    const float* __restrict__ b2_t1, const float* __restrict__ b2_t2,
    const float* __restrict__ b2_sh,
    const float* __restrict__ gshv, const float* __restrict__ g1v,
    const float* __restrict__ g2v,
    float* __restrict__ out_sh, float* __restrict__ out1,
    float* __restrict__ out2) {
  __shared__ unsigned short As[2][128 * LP64];
  __shared__ unsigned short Bs[2][64 * LP64];

  const int tid  = threadIdx.x;
  const int lane = tid & 31;
  const int wave = tid >> 5;
  const int l15  = lane & 15;
  const int half = lane >> 4;
  const int kh   = half * 8;
  const int kb   = half * 16;
  const int wm   = (wave >> 1) * 32;
  const int wn   = (wave & 1) * 32;
  const int m0   = blockIdx.y * 128;
  const int n0   = blockIdx.x * 64;

  v8f ssh0 = {}, ssh1 = {}, ssh2 = {}, ssh3 = {};
  v8f s1_0 = {}, s1_1 = {}, s1_2 = {}, s1_3 = {};
  v8f s2_0 = {}, s2_1 = {}, s2_2 = {}, s2_3 = {};
  v8f c00 = {}, c01 = {}, c10 = {}, c11 = {};

  const int NPH = 12 * 8;   // experts x 64-K phases (HID/64)

  stage128x64(As[0], h_t1 + (size_t)m0 * HID, HID, tid);
  stage64x64(Bs[0], w2_t1 + (size_t)n0 * HID, HID, tid);

  for (int s = 0; s < NPH; ++s) {
    const int ge = s >> 3;
    const int kt = s & 7;
    const int g  = ge >> 2;
    const int e  = ge & 3;
    const int buf = s & 1;

    if (s + 1 < NPH) {
      const int ge2 = (s + 1) >> 3;
      const int kt2 = (s + 1) & 7;
      const int g2  = ge2 >> 2;
      const int e2  = ge2 & 3;
      const unsigned short* hb2 =
          (g2 == 0 ? h_t1 : (g2 == 1 ? h_t2 : h_sh)) +
          (size_t)e2 * B_TOK * HID + (size_t)m0 * HID + kt2 * 64;
      const unsigned short* wb2 =
          (g2 == 0 ? w2_t1 : (g2 == 1 ? w2_t2 : w2_sh)) +
          (size_t)e2 * OUTC * HID + (size_t)n0 * HID + kt2 * 64;
      stage128x64(As[buf ^ 1], hb2, HID, tid);
      stage64x64(Bs[buf ^ 1], wb2, HID, tid);
      wait_async6();
    } else {
      wait_async0();
    }
    __syncthreads();

    const unsigned short* A  = As[buf];
    const unsigned short* Bt = Bs[buf];
    #pragma unroll
    for (int ss = 0; ss < 2; ++ss) {
      const int ko = ss * 32;
      v16bf a0 = lds_a_frag(A + (wm + l15) * LP64 + ko + kb);
      v16bf a1 = lds_a_frag(A + (wm + 16 + l15) * LP64 + ko + kb);
      v16bf b0 = lds_b_frag(Bt + (wn + l15) * LP64 + ko + kb);
      v16bf b1f = lds_b_frag(Bt + (wn + 16 + l15) * LP64 + ko + kb);
      c00 = WMMA_BF16(a0, b0, c00);
      c10 = WMMA_BF16(a1, b0, c10);
      c01 = WMMA_BF16(a0, b1f, c01);
      c11 = WMMA_BF16(a1, b1f, c11);
    }
    __syncthreads();

    if (kt == 7) {
      const float* bb =
          (g == 0 ? b2_t1 : (g == 1 ? b2_t2 : b2_sh)) + (size_t)e * OUTC;
      const int rb0 = m0 + wm, rb1 = m0 + wm + 16;
      const int gc0 = n0 + wn + l15, gc1 = gc0 + 16;
      accum_tile(c00, bb[gc0], gc0, rb0, kh, g, e, ge, gshv, g1v, g2v, ssh0, s1_0, s2_0);
      accum_tile(c10, bb[gc0], gc0, rb1, kh, g, e, ge, gshv, g1v, g2v, ssh1, s1_1, s2_1);
      accum_tile(c01, bb[gc1], gc1, rb0, kh, g, e, ge, gshv, g1v, g2v, ssh2, s1_2, s2_2);
      accum_tile(c11, bb[gc1], gc1, rb1, kh, g, e, ge, gshv, g1v, g2v, ssh3, s1_3, s2_3);
      c00 = (v8f){}; c01 = (v8f){}; c10 = (v8f){}; c11 = (v8f){};
    }
  }

  const int rb0 = m0 + wm, rb1 = m0 + wm + 16;
  const int gc0 = n0 + wn + l15, gc1 = gc0 + 16;
  #pragma unroll
  for (int j = 0; j < 8; ++j) {
    size_t r0 = (size_t)(rb0 + j + kh);
    size_t r1 = (size_t)(rb1 + j + kh);
    out_sh[r0 * OUTC + gc0] = ssh0[j];
    out_sh[r1 * OUTC + gc0] = ssh1[j];
    out_sh[r0 * OUTC + gc1] = ssh2[j];
    out_sh[r1 * OUTC + gc1] = ssh3[j];
    out1[r0 * OUTC + gc0]   = s1_0[j];
    out1[r1 * OUTC + gc0]   = s1_1[j];
    out1[r0 * OUTC + gc1]   = s1_2[j];
    out1[r1 * OUTC + gc1]   = s1_3[j];
    out2[r0 * OUTC + gc0]   = s2_0[j];
    out2[r1 * OUTC + gc0]   = s2_1[j];
    out2[r0 * OUTC + gc1]   = s2_2[j];
    out2[r1 * OUTC + gc1]   = s2_3[j];
  }
}

// ---------------------------------------------------------------------------
// Launch
// ---------------------------------------------------------------------------
extern "C" void kernel_launch(void* const* d_in, const int* in_sizes, int n_in,
                              void* d_out, int out_size, void* d_ws,
                              size_t ws_size, hipStream_t stream) {
  (void)in_sizes; (void)n_in; (void)out_size; (void)ws_size;

  const float* x_full = (const float*)d_in[0];
  const float* x_t1   = (const float*)d_in[1];
  const float* x_t2   = (const float*)d_in[2];
  const float* sh_W1  = (const float*)d_in[3];
  const float* sh_b1  = (const float*)d_in[4];
  const float* sh_W2  = (const float*)d_in[5];
  const float* sh_b2  = (const float*)d_in[6];
  const float* t1_W1  = (const float*)d_in[7];
  const float* t1_b1  = (const float*)d_in[8];
  const float* t1_W2  = (const float*)d_in[9];
  const float* t1_b2  = (const float*)d_in[10];
  const float* t2_W1  = (const float*)d_in[11];
  const float* t2_b1  = (const float*)d_in[12];
  const float* t2_W2  = (const float*)d_in[13];
  const float* t2_b2  = (const float*)d_in[14];
  const float* gsh_W  = (const float*)d_in[15];
  const float* gsh_b  = (const float*)d_in[16];
  const float* g1_W   = (const float*)d_in[17];
  const float* g1_b   = (const float*)d_in[18];
  const float* g2_W   = (const float*)d_in[19];
  const float* g2_b   = (const float*)d_in[20];

  // Workspace layout (byte offsets)
  char* ws = (char*)d_ws;
  unsigned short* xf_bf  = (unsigned short*)(ws + 0);         //  8,388,608
  unsigned short* x1_bf  = (unsigned short*)(ws + 8388608);   //  4,194,304
  unsigned short* x2_bf  = (unsigned short*)(ws + 12582912);  //  4,194,304
  unsigned short* w1s_t  = (unsigned short*)(ws + 16777216);  //  4,194,304 [4,512,1024]
  unsigned short* w1a_t  = (unsigned short*)(ws + 20971520);  //  2,097,152 [4,512,512]
  unsigned short* w1b_t  = (unsigned short*)(ws + 23068672);  //  2,097,152
  unsigned short* w2s_t  = (unsigned short*)(ws + 25165824);  //  1,048,576 [4,256,512]
  unsigned short* w2a_t  = (unsigned short*)(ws + 26214400);  //  1,048,576
  unsigned short* w2b_t  = (unsigned short*)(ws + 27262976);  //  1,048,576
  unsigned short* h_sh   = (unsigned short*)(ws + 28311552);  // 16,777,216 [4,4096,512]
  unsigned short* h_t1   = (unsigned short*)(ws + 45088768);  // 16,777,216
  unsigned short* h_t2   = (unsigned short*)(ws + 61865984);  // 16,777,216
  float*          gsh_v  = (float*)(ws + 78643200);           //    196,608 [4096,12]
  float*          g1_v   = (float*)(ws + 78839808);           //    131,072 [4096,8]
  float*          g2_v   = (float*)(ws + 78970880);           //    131,072

  float* out_sh = (float*)d_out;
  float* out1   = out_sh + (size_t)B_TOK * OUTC;
  float* out2   = out1 + (size_t)B_TOK * OUTC;

  const int TPB = 256;
  #define NB(n) (((n) + TPB - 1) / TPB)

  // 1) Convert activations to bf16 (8 elems/thread)
  cvt_bf16_kernel<<<NB(B_TOK * D_FULL / 8), TPB, 0, stream>>>(x_full, xf_bf, B_TOK * D_FULL / 8);
  cvt_bf16_kernel<<<NB(B_TOK * D_TASK / 8), TPB, 0, stream>>>(x_t1, x1_bf, B_TOK * D_TASK / 8);
  cvt_bf16_kernel<<<NB(B_TOK * D_TASK / 8), TPB, 0, stream>>>(x_t2, x2_bf, B_TOK * D_TASK / 8);

  // 2) Convert + transpose weights: W1 [E,D,H]->[E,H,D], W2 [E,H,O]->[E,O,H]
  cvt_tr_bf16_kernel<<<NB(NEXP * D_FULL * HID / 8), TPB, 0, stream>>>(sh_W1, w1s_t, NEXP, D_FULL, HID);
  cvt_tr_bf16_kernel<<<NB(NEXP * D_TASK * HID / 8), TPB, 0, stream>>>(t1_W1, w1a_t, NEXP, D_TASK, HID);
  cvt_tr_bf16_kernel<<<NB(NEXP * D_TASK * HID / 8), TPB, 0, stream>>>(t2_W1, w1b_t, NEXP, D_TASK, HID);
  cvt_tr_bf16_kernel<<<NB(NEXP * HID * OUTC / 8), TPB, 0, stream>>>(sh_W2, w2s_t, NEXP, HID, OUTC);
  cvt_tr_bf16_kernel<<<NB(NEXP * HID * OUTC / 8), TPB, 0, stream>>>(t1_W2, w2a_t, NEXP, HID, OUTC);
  cvt_tr_bf16_kernel<<<NB(NEXP * HID * OUTC / 8), TPB, 0, stream>>>(t2_W2, w2b_t, NEXP, HID, OUTC);

  // 3) Gates: one wave per token
  gates_kernel<<<NB(B_TOK * 32), TPB, 0, stream>>>(x_full, x_t1, x_t2,
                                                   gsh_W, gsh_b, g1_W, g1_b,
                                                   g2_W, g2_b, gsh_v, g1_v, g2_v);

  // 4) Layer-1 expert GEMMs (WMMA bf16, async-LDS double buffered)
  dim3 g1grid(HID / 128, B_TOK / 128, NEXP);
  expert_gemm1_kernel<<<g1grid, TPB, 0, stream>>>(xf_bf, w1s_t, sh_b1, h_sh, D_FULL);
  expert_gemm1_kernel<<<g1grid, TPB, 0, stream>>>(x1_bf, w1a_t, t1_b1, h_t1, D_TASK);
  expert_gemm1_kernel<<<g1grid, TPB, 0, stream>>>(x2_bf, w1b_t, t2_b1, h_t2, D_TASK);

  // 5) Layer-2 GEMM fused with gated combine (WMMA bf16, deterministic)
  dim3 g2grid(OUTC / 64, B_TOK / 128);
  moe_gemm2_combine_kernel<<<g2grid, TPB, 0, stream>>>(
      h_t1, h_t2, h_sh, w2a_t, w2b_t, w2s_t, t1_b2, t2_b2, sh_b2,
      gsh_v, g1_v, g2_v, out_sh, out1, out2);

  #undef NB
}
